// PointNet2Ind_7473243095567
// MI455X (gfx1250) — matrix operand order repaired and checked by reading, not compiled, not run on hardware
//
#include <hip/hip_runtime.h>
#include <hip/hip_bf16.h>

typedef __attribute__((ext_vector_type(16))) _Float16 v16h;
typedef __attribute__((ext_vector_type(8)))  _Float16 v8h;
typedef __attribute__((ext_vector_type(8)))  float    v8f;

// ---------------------------------------------------------------- utilities
__global__ void copy_f32_kernel(const float* __restrict__ s, float* __restrict__ d, int n) {
  int t = blockIdx.x * blockDim.x + threadIdx.x;
  if (t < n) d[t] = s[t];
}

__global__ void seed_copy_kernel(const int* __restrict__ src, int* __restrict__ dst) {
  int t = blockIdx.x * blockDim.x + threadIdx.x;
  if (t < 512) dst[t] = src[(t / 128) * 512 + (t % 128)];
}

// W (K x N, f32 row-major) -> Wt (N x Kpad, f16, zero-padded K)
__global__ void wconv_kernel(const float* __restrict__ W, _Float16* __restrict__ Wt,
                             int K, int N, int Kpad) {
  int t = blockIdx.x * blockDim.x + threadIdx.x;
  if (t >= N * Kpad) return;
  int n = t / Kpad, k = t % Kpad;
  float v = (k < K) ? W[(size_t)k * N + n] : 0.0f;
  Wt[t] = (_Float16)v;
}

// ---------------------------------------------------------------- FPS
#define FPS_T 256
__global__ void fps_kernel(const float* __restrict__ xyz,  // B x N x 3
                           float* __restrict__ dist,       // B x N scratch
                           int* __restrict__ inds,         // B x np
                           float* __restrict__ newxyz,     // B x np x 3
                           int N, int np) {
  const int b = blockIdx.x;
  const int t = threadIdx.x;
  const float* P = xyz + (size_t)b * N * 3;
  float* D = dist + (size_t)b * N;
  __shared__ float s_v[FPS_T];
  __shared__ int   s_i[FPS_T];
  __shared__ float s_c[3];
  __shared__ int   s_far;
  for (int p = t; p < N; p += FPS_T) D[p] = 1e10f;
  if (t == 0) s_far = 0;
  __syncthreads();
  for (int i = 0; i < np; ++i) {
    if (t == 0) {
      int far = s_far;
      inds[b * np + i] = far;
      float cx = P[far * 3 + 0], cy = P[far * 3 + 1], cz = P[far * 3 + 2];
      s_c[0] = cx; s_c[1] = cy; s_c[2] = cz;
      newxyz[((size_t)b * np + i) * 3 + 0] = cx;
      newxyz[((size_t)b * np + i) * 3 + 1] = cy;
      newxyz[((size_t)b * np + i) * 3 + 2] = cz;
    }
    __syncthreads();
    float cx = s_c[0], cy = s_c[1], cz = s_c[2];
    float best = -1.0f; int bi = 0;
    for (int p = t; p < N; p += FPS_T) {
      float dx = P[p * 3 + 0] - cx;
      float dy = P[p * 3 + 1] - cy;
      float dz = P[p * 3 + 2] - cz;
      float d = dx * dx + dy * dy + dz * dz;
      float nd = fminf(D[p], d);
      D[p] = nd;
      if (nd > best) { best = nd; bi = p; }   // '>' keeps first index on ties
    }
    s_v[t] = best; s_i[t] = bi;
    __syncthreads();
    for (int s = FPS_T / 2; s > 0; s >>= 1) {
      if (t < s) {
        float ov = s_v[t + s]; int oi = s_i[t + s];
        if (ov > s_v[t] || (ov == s_v[t] && oi < s_i[t])) { s_v[t] = ov; s_i[t] = oi; }
      }
      __syncthreads();
    }
    if (t == 0) s_far = s_i[0];
    __syncthreads();
  }
}

// ---------------------------------------------------------------- ball query
// one wave32 per centroid; collects the first 32 in-ball indices in ascending
// index order (== sort-then-truncate in the reference), fills with first.
__global__ void ball_query_kernel(const float* __restrict__ xyz,   // B x N x 3
                                  const float* __restrict__ nxyz,  // B x np x 3
                                  int* __restrict__ idx,           // B x np x 32
                                  int N, int np, float r2) {
  const int b = blockIdx.x / np;
  const int j = blockIdx.x % np;
  const int lane = threadIdx.x;
  const float* P = xyz + (size_t)b * N * 3;
  const float cx = nxyz[((size_t)b * np + j) * 3 + 0];
  const float cy = nxyz[((size_t)b * np + j) * 3 + 1];
  const float cz = nxyz[((size_t)b * np + j) * 3 + 2];
  int* out = idx + ((size_t)b * np + j) * 32;
  __shared__ int s_cnt;
  if (lane == 0) s_cnt = 0;
  __syncthreads();
  const int chunks = (N + 31) / 32;
  for (int c = 0; c < chunks; ++c) {
    int p = c * 32 + lane;
    bool inb = false;
    if (p < N) {
      float dx = P[p * 3 + 0] - cx;
      float dy = P[p * 3 + 1] - cy;
      float dz = P[p * 3 + 2] - cz;
      inb = (dx * dx + dy * dy + dz * dz) <= r2;
    }
    unsigned long long m = __ballot(inb);
    if (lane == 0) {
      int cnt = s_cnt;
      unsigned int mm = (unsigned int)m;
      while (mm && cnt < 32) {
        int bit = __builtin_ctz(mm);
        mm &= mm - 1;
        out[cnt++] = c * 32 + bit;
      }
      s_cnt = cnt;
    }
    __syncthreads();
    if (s_cnt >= 32) break;
    __syncthreads();
  }
  __syncthreads();
  int cnt = s_cnt;
  int first = 0;
  if (cnt > 0) first = out[0];
  for (int s = cnt + lane; s < 32; s += 32) out[s] = first;
}

// ---------------------------------------------------------------- grouping
// X[row, k] row = ((b*np)+j)*32+s : k<3 -> (xyz[idx]-new)/r ; k<3+C -> feat ; else 0
__global__ void group_build_kernel(const float* __restrict__ xyz,      // B x Nin x 3
                                   const float* __restrict__ nxyz,     // B x np x 3
                                   const _Float16* __restrict__ feat,  // B x Nin x C (or null)
                                   const int* __restrict__ idx,        // B x np x 32
                                   _Float16* __restrict__ X,           // (B*np*32) x Kpad
                                   int Nin, int np, int C, int Kpad, float inv_r,
                                   long long total) {
  long long t = (long long)blockIdx.x * blockDim.x + threadIdx.x;
  if (t >= total) return;
  int k = (int)(t % Kpad);
  long long row = t / Kpad;
  int s = (int)(row & 31);
  long long bj = row >> 5;          // b*np + j
  int b = (int)(bj / np);
  int pid = idx[bj * 32 + s];
  float v = 0.0f;
  if (k < 3) {
    v = (xyz[((size_t)b * Nin + pid) * 3 + k] - nxyz[bj * 3 + k]) * inv_r;
  } else if (k < 3 + C) {
    v = (float)feat[((size_t)b * Nin + pid) * C + (k - 3)];
  }
  X[t] = (_Float16)v;
}

// ---------------------------------------------------------------- WMMA GEMM
// Y[M,N] = relu(A[M,Ka] * Wt[N,Ka]^T + bias), f16 A/W, f32 accumulate.
// Each wave owns a 32x32 register tile (2x2 WMMA fragments, A/B each reused
// twice -> 16 FLOP/byte from L2 instead of 8). 4 waves/block in a 2x2
// arrangement cover a 64x64 block tile. M,N are multiples of 64 by padding.
__global__ __launch_bounds__(128)
void gemm_wmma_kernel(const _Float16* __restrict__ A,   // M x Ka
                      const _Float16* __restrict__ Wt,  // N x Ka (pre-transposed)
                      const float* __restrict__ bias,   // N
                      _Float16* __restrict__ Yh,        // M x N (nullable)
                      float* __restrict__ Yf,           // M x N (nullable)
                      int M, int N, int Ka, int relu) {
  const int lane = threadIdx.x & 31;
  const int wave = threadIdx.x >> 5;
  const int m0 = blockIdx.x * 64 + (wave >> 1) * 32;
  const int n0 = blockIdx.y * 64 + (wave & 1) * 32;
  const int half_sel = lane >> 4;          // lanes 0-15 vs 16-31
  const int l15 = lane & 15;

  // A: M = lane&15 within each 16-row fragment; B: N = lane&15 likewise.
  const _Float16* arow0 = A  + (size_t)(m0 + l15) * Ka;
  const _Float16* arow1 = A  + (size_t)(m0 + 16 + l15) * Ka;
  const _Float16* brow0 = Wt + (size_t)(n0 + l15) * Ka;
  const _Float16* brow1 = Wt + (size_t)(n0 + 16 + l15) * Ka;

  union { v16h v; v8h h[2]; } af0, af1, bf0, bf1;
  v8f acc00 = {}, acc01 = {}, acc10 = {}, acc11 = {};
  for (int kt = 0; kt < Ka; kt += 32) {
    // A 16x32 f16: lanes 0-15 K=kt..kt+15,
    //              lanes 16-31 K=kt+16..kt+31
    af0.h[0] = *(const v8h*)(arow0 + kt + half_sel * 16);
    af0.h[1] = *(const v8h*)(arow0 + kt + half_sel * 16 + 8);
    af1.h[0] = *(const v8h*)(arow1 + kt + half_sel * 16);
    af1.h[1] = *(const v8h*)(arow1 + kt + half_sel * 16 + 8);
    // B 32x16 f16: lanes 0-15 K=kt..kt+15, lanes 16-31 K=kt+16..kt+31
    bf0.v = *(const v16h*)(brow0 + kt + half_sel * 16);
    bf1.v = *(const v16h*)(brow1 + kt + half_sel * 16);
    if (kt + 32 < Ka) {
      __builtin_prefetch(arow0 + kt + 32, 0, 3);   // global_prefetch_b8 (near)
      __builtin_prefetch(arow1 + kt + 32, 0, 3);
      __builtin_prefetch(brow0 + kt + 32, 0, 3);
      __builtin_prefetch(brow1 + kt + 32, 0, 3);
    }
    acc00 = __builtin_amdgcn_wmma_f32_16x16x32_f16(false, af0.v, false, bf0.v,
                                                   (short)0, acc00, false, false);
    acc01 = __builtin_amdgcn_wmma_f32_16x16x32_f16(false, af0.v, false, bf1.v,
                                                   (short)0, acc01, false, false);
    acc10 = __builtin_amdgcn_wmma_f32_16x16x32_f16(false, af1.v, false, bf0.v,
                                                   (short)0, acc10, false, false);
    acc11 = __builtin_amdgcn_wmma_f32_16x16x32_f16(false, af1.v, false, bf1.v,
                                                   (short)0, acc11, false, false);
  }
  // C/D f32 16x16: VGPR r -> M = tile_m0 + half_sel*8 + r, N = tile_n0 + (lane&15)
  const int col0 = n0 + l15;
  const int col1 = n0 + 16 + l15;
  const float bv0 = bias[col0];
  const float bv1 = bias[col1];
  for (int r = 0; r < 8; ++r) {
    int row0 = m0 + half_sel * 8 + r;
    int row1 = row0 + 16;
    float v00 = acc00[r] + bv0;
    float v01 = acc01[r] + bv1;
    float v10 = acc10[r] + bv0;
    float v11 = acc11[r] + bv1;
    if (relu) {
      v00 = fmaxf(v00, 0.0f); v01 = fmaxf(v01, 0.0f);
      v10 = fmaxf(v10, 0.0f); v11 = fmaxf(v11, 0.0f);
    }
    size_t o00 = (size_t)row0 * N + col0, o01 = (size_t)row0 * N + col1;
    size_t o10 = (size_t)row1 * N + col0, o11 = (size_t)row1 * N + col1;
    if (Yh) {
      Yh[o00] = (_Float16)v00; Yh[o01] = (_Float16)v01;
      Yh[o10] = (_Float16)v10; Yh[o11] = (_Float16)v11;
    }
    if (Yf) {
      Yf[o00] = v00; Yf[o01] = v01;
      Yf[o10] = v10; Yf[o11] = v11;
    }
  }
}

// ---------------------------------------------------------------- maxpool over 32 samples
__global__ void maxpool_kernel(const _Float16* __restrict__ Y,  // (Bnp*32) x N
                               float* __restrict__ Ff,          // Bnp x N
                               _Float16* __restrict__ Fh,       // Bnp x N (nullable)
                               int Bnp, int N) {
  int t = blockIdx.x * blockDim.x + threadIdx.x;
  if (t >= Bnp * N) return;
  int pt = t / N, ch = t % N;
  const _Float16* src = Y + ((size_t)pt * 32) * N + ch;
  float m = -1e30f;
  for (int s = 0; s < 32; ++s) m = fmaxf(m, (float)src[(size_t)s * N]);
  Ff[t] = m;
  if (Fh) Fh[t] = (_Float16)m;
}

// ---------------------------------------------------------------- 3-NN (top_k of -d2)
__global__ void knn3_kernel(const float* __restrict__ U,   // B x Nu x 3
                            const float* __restrict__ Kn,  // B x Nk x 3
                            int* __restrict__ oidx,        // (B*Nu) x 3
                            float* __restrict__ ow,        // (B*Nu) x 3
                            int Nu, int Nk, int total) {
  int t = blockIdx.x * blockDim.x + threadIdx.x;
  if (t >= total) return;
  int b = t / Nu;
  float ux = U[(size_t)t * 3 + 0], uy = U[(size_t)t * 3 + 1], uz = U[(size_t)t * 3 + 2];
  const float* P = Kn + (size_t)b * Nk * 3;
  float d0 = 1e30f, d1 = 1e30f, d2 = 1e30f;
  int i0 = 0, i1 = 0, i2 = 0;
  for (int p = 0; p < Nk; ++p) {
    float dx = P[p * 3 + 0] - ux;
    float dy = P[p * 3 + 1] - uy;
    float dz = P[p * 3 + 2] - uz;
    float d = dx * dx + dy * dy + dz * dz;
    if (d < d0)      { d2 = d1; i2 = i1; d1 = d0; i1 = i0; d0 = d; i0 = p; }
    else if (d < d1) { d2 = d1; i2 = i1; d1 = d;  i1 = p; }
    else if (d < d2) { d2 = d;  i2 = p; }
  }
  float w0 = 1.0f / (d0 + 1e-8f), w1 = 1.0f / (d1 + 1e-8f), w2 = 1.0f / (d2 + 1e-8f);
  float ws = w0 + w1 + w2;
  oidx[t * 3 + 0] = i0; oidx[t * 3 + 1] = i1; oidx[t * 3 + 2] = i2;
  ow[t * 3 + 0] = w0 / ws; ow[t * 3 + 1] = w1 / ws; ow[t * 3 + 2] = w2 / ws;
}

// X[row, k] = k<Cf ? sum_j w*Fk[idx_j,k] : Fu[row, k-Cf]   (f16 out)
__global__ void interp_build_kernel(const int* __restrict__ idx3,   // (B*Nu) x 3
                                    const float* __restrict__ w3,   // (B*Nu) x 3
                                    const float* __restrict__ Fk,   // B x Nk x Cf
                                    const float* __restrict__ Fu,   // B x Nu x Cu (nullable)
                                    _Float16* __restrict__ X,       // (B*Nu) x (Cf+Cu)
                                    int Nu, int Nk, int Cf, int Cu,
                                    long long total) {
  long long t = (long long)blockIdx.x * blockDim.x + threadIdx.x;
  if (t >= total) return;
  int Ktot = Cf + Cu;
  long long row = t / Ktot;
  int k = (int)(t % Ktot);
  int b = (int)(row / Nu);
  float v;
  if (k < Cf) {
    const float* F = Fk + (size_t)b * Nk * Cf;
    v = w3[row * 3 + 0] * F[(size_t)idx3[row * 3 + 0] * Cf + k]
      + w3[row * 3 + 1] * F[(size_t)idx3[row * 3 + 1] * Cf + k]
      + w3[row * 3 + 2] * F[(size_t)idx3[row * 3 + 2] * Cf + k];
  } else {
    v = Fu[row * (size_t)Cu + (k - Cf)];
  }
  X[t] = (_Float16)v;
}

// ================================================================ host
extern "C" void kernel_launch(void* const* d_in, const int* in_sizes, int n_in,
                              void* d_out, int out_size, void* d_ws, size_t ws_size,
                              hipStream_t stream) {
  (void)in_sizes; (void)n_in; (void)out_size; (void)ws_size;
  char* ws = (char*)d_ws;
  size_t cur = 0;
  auto alloc = [&](size_t bytes) -> char* {
    char* p = ws + cur;
    cur += (bytes + 255) & ~(size_t)255;
    return p;
  };
  auto cdiv = [](long long a, long long b) { return (int)((a + b - 1) / b); };

  // layer table: {weight input index, K, N, Kpad}; bias = d_in[widx+1]
  struct LayerDef { int widx, K, N, Kpad; };
  const LayerDef L[16] = {
    {1, 3, 64, 32},    {3, 64, 64, 64},    {5, 64, 128, 64},     // sa1
    {7, 131, 128, 160},{9, 128, 128, 128}, {11, 128, 256, 128},  // sa2
    {13, 259, 256, 288},{15, 256, 256, 256},{17, 256, 256, 256}, // sa3
    {19, 256, 256, 256},                                         // trans
    {21, 512, 128, 512},{23, 128, 128, 128},                     // fp1
    {25, 256, 256, 256},{27, 256, 256, 256},                     // fp2
    {29, 256, 256, 256},{31, 256, 256, 256},                     // fp3
  };
  _Float16* Wt[16];
  for (int i = 0; i < 16; ++i) Wt[i] = (_Float16*)alloc((size_t)L[i].N * L[i].Kpad * 2);

  float*    dist    = (float*)alloc((size_t)4 * 16384 * 4);
  int*      inds1   = (int*)alloc((size_t)4 * 512 * 4);
  int*      inds2   = (int*)alloc((size_t)4 * 256 * 4);
  int*      inds3   = (int*)alloc((size_t)4 * 128 * 4);
  float*    sa1_xyz = (float*)alloc((size_t)4 * 512 * 3 * 4);
  float*    sa2_xyz = (float*)alloc((size_t)4 * 256 * 3 * 4);
  float*    sa3_xyz = (float*)alloc((size_t)4 * 128 * 3 * 4);
  int*      gidx1   = (int*)alloc((size_t)4 * 512 * 32 * 4);
  int*      gidx2   = (int*)alloc((size_t)4 * 256 * 32 * 4);
  int*      gidx3   = (int*)alloc((size_t)4 * 128 * 32 * 4);
  float*    sa1_f   = (float*)alloc((size_t)4 * 512 * 128 * 4);
  _Float16* sa1_fh  = (_Float16*)alloc((size_t)4 * 512 * 128 * 2);
  float*    sa2_f   = (float*)alloc((size_t)4 * 256 * 256 * 4);
  _Float16* sa2_fh  = (_Float16*)alloc((size_t)4 * 256 * 256 * 2);
  float*    sa3_f   = (float*)alloc((size_t)4 * 128 * 256 * 4);
  _Float16* sa3_fh  = (_Float16*)alloc((size_t)4 * 128 * 256 * 2);
  float*    fp1f    = (float*)alloc((size_t)4 * 256 * 128 * 4);
  float*    fp2f    = (float*)alloc((size_t)4 * 512 * 256 * 4);
  int*      kidx    = (int*)alloc((size_t)4 * 16384 * 3 * 4);
  float*    kw      = (float*)alloc((size_t)4 * 16384 * 3 * 4);
  _Float16* ping    = (_Float16*)alloc((size_t)65536 * 256 * 2);
  _Float16* pong    = (_Float16*)alloc((size_t)65536 * 256 * 2);

  const float* pc = (const float*)d_in[0];
  float* out      = (float*)d_out;
  float* out_fp3  = out + 196608;
  float* out_s3x  = out + 16973824;
  float* out_mid  = out + 16975360;
  int*   out_seed = (int*)(out + 17106432);

  auto gemm = [&](const _Float16* A, int li, _Float16* Yh, float* Yf,
                  int M, int N, int Ka, int relu) {
    dim3 grid(M / 64, N / 64);
    gemm_wmma_kernel<<<grid, 128, 0, stream>>>(
        A, Wt[li], (const float*)d_in[L[li].widx + 1], Yh, Yf, M, N, Ka, relu);
  };

  // weight prep (f32 KxN -> f16 NxKpad)
  for (int i = 0; i < 16; ++i) {
    int tot = L[i].N * L[i].Kpad;
    wconv_kernel<<<cdiv(tot, 256), 256, 0, stream>>>(
        (const float*)d_in[L[i].widx], Wt[i], L[i].K, L[i].N, L[i].Kpad);
  }

  // output 0: xyz copy
  copy_f32_kernel<<<cdiv(196608, 256), 256, 0, stream>>>(pc, out, 196608);

  // ---------------- SA1: 16384 -> 512, r=0.3, [3,64,64,128]
  fps_kernel<<<4, 256, 0, stream>>>(pc, dist, inds1, sa1_xyz, 16384, 512);
  seed_copy_kernel<<<2, 256, 0, stream>>>(inds1, out_seed);
  ball_query_kernel<<<4 * 512, 32, 0, stream>>>(pc, sa1_xyz, gidx1, 16384, 512, 0.09f);
  {
    long long tot = (long long)4 * 512 * 32 * 32;
    group_build_kernel<<<cdiv(tot, 256), 256, 0, stream>>>(
        pc, sa1_xyz, (const _Float16*)nullptr, gidx1, ping, 16384, 512, 0, 32, 1.0f / 0.3f, tot);
  }
  gemm(ping, 0, pong, nullptr, 65536, 64, 32, 1);
  gemm(pong, 1, ping, nullptr, 65536, 64, 64, 1);
  gemm(ping, 2, pong, nullptr, 65536, 128, 64, 1);
  maxpool_kernel<<<cdiv(2048 * 128, 256), 256, 0, stream>>>(pong, sa1_f, sa1_fh, 2048, 128);

  // ---------------- SA2: 512 -> 256, r=0.5, [131,128,128,256]
  fps_kernel<<<4, 256, 0, stream>>>(sa1_xyz, dist, inds2, sa2_xyz, 512, 256);
  ball_query_kernel<<<4 * 256, 32, 0, stream>>>(sa1_xyz, sa2_xyz, gidx2, 512, 256, 0.25f);
  {
    long long tot = (long long)4 * 256 * 32 * 160;
    group_build_kernel<<<cdiv(tot, 256), 256, 0, stream>>>(
        sa1_xyz, sa2_xyz, sa1_fh, gidx2, ping, 512, 256, 128, 160, 1.0f / 0.5f, tot);
  }
  gemm(ping, 3, pong, nullptr, 32768, 128, 160, 1);
  gemm(pong, 4, ping, nullptr, 32768, 128, 128, 1);
  gemm(ping, 5, pong, nullptr, 32768, 256, 128, 1);
  maxpool_kernel<<<cdiv(1024 * 256, 256), 256, 0, stream>>>(pong, sa2_f, sa2_fh, 1024, 256);

  // ---------------- SA3: 256 -> 128, r=0.7, [259,256,256,256]
  fps_kernel<<<4, 256, 0, stream>>>(sa2_xyz, dist, inds3, sa3_xyz, 256, 128);
  copy_f32_kernel<<<cdiv(1536, 256), 256, 0, stream>>>(sa3_xyz, out_s3x, 1536);
  ball_query_kernel<<<4 * 128, 32, 0, stream>>>(sa2_xyz, sa3_xyz, gidx3, 256, 128, 0.49f);
  {
    long long tot = (long long)4 * 128 * 32 * 288;
    group_build_kernel<<<cdiv(tot, 256), 256, 0, stream>>>(
        sa2_xyz, sa3_xyz, sa2_fh, gidx3, ping, 256, 128, 256, 288, 1.0f / 0.7f, tot);
  }
  gemm(ping, 6, pong, nullptr, 16384, 256, 288, 1);
  gemm(pong, 7, ping, nullptr, 16384, 256, 256, 1);
  gemm(ping, 8, pong, nullptr, 16384, 256, 256, 1);
  maxpool_kernel<<<cdiv(512 * 256, 256), 256, 0, stream>>>(pong, sa3_f, sa3_fh, 512, 256);

  // ---------------- mid_features = sa3_f @ trans_W + trans_b  (no relu)
  gemm(sa3_fh, 9, nullptr, out_mid, 512, 256, 256, 0);

  // ---------------- FP1: known sa3 (128) -> unknown sa2 (256), [512,128,128]
  knn3_kernel<<<cdiv(1024, 256), 256, 0, stream>>>(sa2_xyz, sa3_xyz, kidx, kw, 256, 128, 1024);
  {
    long long tot = (long long)4 * 256 * 512;
    interp_build_kernel<<<cdiv(tot, 256), 256, 0, stream>>>(
        kidx, kw, sa3_f, sa2_f, ping, 256, 128, 256, 256, tot);
  }
  gemm(ping, 10, pong, nullptr, 1024, 128, 512, 1);
  gemm(pong, 11, nullptr, fp1f, 1024, 128, 128, 1);

  // ---------------- FP2: known sa2 (256) -> unknown sa1 (512), [256,256,256]
  knn3_kernel<<<cdiv(2048, 256), 256, 0, stream>>>(sa1_xyz, sa2_xyz, kidx, kw, 512, 256, 2048);
  {
    long long tot = (long long)4 * 512 * 256;
    interp_build_kernel<<<cdiv(tot, 256), 256, 0, stream>>>(
        kidx, kw, fp1f, sa1_f, ping, 512, 256, 128, 128, tot);
  }
  gemm(ping, 12, pong, nullptr, 2048, 256, 256, 1);
  gemm(pong, 13, nullptr, fp2f, 2048, 256, 256, 1);

  // ---------------- FP3: known sa1 (512) -> unknown xyz (16384), [256,256,256]
  knn3_kernel<<<cdiv(65536, 256), 256, 0, stream>>>(pc, sa1_xyz, kidx, kw, 16384, 512, 65536);
  {
    long long tot = (long long)4 * 16384 * 256;
    interp_build_kernel<<<cdiv(tot, 256), 256, 0, stream>>>(
        kidx, kw, fp2f, (const float*)nullptr, ping, 16384, 512, 256, 0, tot);
  }
  gemm(ping, 14, pong, nullptr, 65536, 256, 256, 1);
  gemm(pong, 15, nullptr, out_fp3, 65536, 256, 256, 1);
}
